// CeilingSymbolicRNN_79886391705968
// MI455X (gfx1250) — compile-verified, matrix operand order checked by reading, not run on hardware
//
#include <hip/hip_runtime.h>
#include <math.h>
#include <stdint.h>

// ---------------------------------------------------------------------------
// CeilingSymbolicRNN for MI455X (gfx1250, wave32, WMMA + async global->LDS)
// ---------------------------------------------------------------------------

#define B_ 32
#define S_ 512
#define V_ 256
#define D_ 512
#define TWO_D 1024
#define M_TOT (B_ * S_)        // 16384 rows
#define NSYM_ 512
#define NCON_ 64
#define DEPTH_ 6
#define LOOK_ 3
#define MEMSPAN_ 10
#define CC_ 0.25f
#define EPS_ 1e-8f

typedef __bf16 bf16;
typedef __attribute__((ext_vector_type(16))) __bf16 v16bf;
typedef __attribute__((ext_vector_type(8)))  __bf16 v8bf;
typedef __attribute__((ext_vector_type(8)))  float  v8f;

union Frag { v16bf v; struct { v8bf lo, hi; } p; };

// ---------------- CDNA5 async global->LDS (ASYNCcnt) ----------------
// GVS form: mem_addr = SGPR64 base + VGPR32 byte offset; VDST = LDS byte addr.
__device__ __forceinline__ void async_ld_b128(uint32_t lds_addr,
                                              uint32_t voff_bytes,
                                              const void* sbase) {
  asm volatile("global_load_async_to_lds_b128 %0, %1, %2"
               :: "v"(lds_addr), "v"(voff_bytes), "s"(sbase)
               : "memory");
}
__device__ __forceinline__ void wait_async0() {
  asm volatile("s_wait_asynccnt 0x0" ::: "memory");
}

// ---------------- wave32 reduction helpers ----------------
__device__ __forceinline__ float waveSum(float v) {
  #pragma unroll
  for (int off = 16; off > 0; off >>= 1) v += __shfl_xor(v, off, 32);
  return v;
}
__device__ __forceinline__ void waveArgMin(float& d, int& i) {
  #pragma unroll
  for (int off = 16; off > 0; off >>= 1) {
    float od = __shfl_xor(d, off, 32);
    int   oi = __shfl_xor(i, off, 32);
    if (od < d || (od == d && oi < i)) { d = od; i = oi; }
  }
}

// ---------------------------------------------------------------------------
// GEMM: C(MxN,f32 or bf16) = A_bf16(MxK) @ W_bf16(NxK)^T  [+ bias(N)]
// Requires M%128==0, N%64==0, K%64==0 (always true here).
// Block tile 128x64, 8 waves, each wave 32x32 via 2x2 WMMA accumulators.
// Double-buffered LDS filled by global_load_async_to_lds_b128 (ASYNCcnt).
// ---------------------------------------------------------------------------
#define TILE_M 128
#define TILE_N 64
#define TILE_K 64
#define LDA (TILE_K + 8)   // bf16 stride 72 -> 144B rows, keeps 16B alignment

template<bool STORE_BF16>
__global__ __launch_bounds__(256)
void gemm_bf16_kernel(const bf16* __restrict__ A, const bf16* __restrict__ W,
                      const float* __restrict__ bias, void* __restrict__ Cout,
                      int M, int N, int K)
{
  __shared__ __align__(16) bf16 Alds[2][TILE_M][LDA];
  __shared__ __align__(16) bf16 Wlds[2][TILE_N][LDA];

  const int tid  = threadIdx.x;
  const int lane = tid & 31;
  const int wid  = tid >> 5;
  const int wm   = wid & 3;        // 4 M-groups of 32 rows
  const int wn   = wid >> 2;       // 2 N-groups of 32 cols
  const int m0   = blockIdx.y * TILE_M;
  const int n0   = blockIdx.x * TILE_N;

  const int half = lane >> 4;      // selects K sub-range per ISA A/B layout
  const int r16  = lane & 15;

  // this thread's staging chunks (fixed across K-tiles)
  const int cA_row[4] = { (tid + 0)   >> 3, (tid + 256) >> 3,
                          (tid + 512) >> 3, (tid + 768) >> 3 };
  const int cW_row[2] = { (tid + 0) >> 3, (tid + 256) >> 3 };
  const int c8 = (tid & 7) * 8;    // column chunk (8 bf16 = 16B)

  auto stage = [&](int k0, int buf) {
    #pragma unroll
    for (int i = 0; i < 4; ++i) {
      uint32_t voff = ((uint32_t)(m0 + cA_row[i]) * (uint32_t)K + k0 + c8) * 2u;
      async_ld_b128((uint32_t)(uintptr_t)&Alds[buf][cA_row[i]][c8], voff, A);
    }
    #pragma unroll
    for (int i = 0; i < 2; ++i) {
      uint32_t voff = ((uint32_t)(n0 + cW_row[i]) * (uint32_t)K + k0 + c8) * 2u;
      async_ld_b128((uint32_t)(uintptr_t)&Wlds[buf][cW_row[i]][c8], voff, W);
    }
  };

  v8f acc[2][2];
  const v8f vzero = {0.f,0.f,0.f,0.f,0.f,0.f,0.f,0.f};
  #pragma unroll
  for (int i = 0; i < 2; ++i)
    #pragma unroll
    for (int j = 0; j < 2; ++j) acc[i][j] = vzero;

  // prologue: fill buffer 0
  stage(0, 0);
  wait_async0();
  __syncthreads();

  int cur = 0;
  for (int k0 = 0; k0 < K; k0 += TILE_K) {
    if (k0 + TILE_K < K) {
      stage(k0 + TILE_K, cur ^ 1);              // overlap copy with compute
      if (k0 + 2 * TILE_K < K)                  // L2 warm-up two tiles ahead
        __builtin_prefetch(&A[(size_t)(m0 + (tid >> 3)) * K + k0 + 2 * TILE_K], 0, 1);
    }

    #pragma unroll
    for (int kk = 0; kk < TILE_K; kk += 32) {
      Frag a[2], b[2];
      #pragma unroll
      for (int i = 0; i < 2; ++i) {
        // 16-bit A layout: lanes 0-15 hold K {0..7,16..23}, lanes 16-31 hold
        // K {8..15,24..31}; row M = lane&15. B mirrors with column n.
        const bf16* pa = &Alds[cur][wm * 32 + i * 16 + r16][kk + half * 8];
        a[i].p.lo = *(const v8bf*)pa;
        a[i].p.hi = *(const v8bf*)(pa + 16);
        const bf16* pb = &Wlds[cur][wn * 32 + i * 16 + r16][kk + half * 8];
        b[i].p.lo = *(const v8bf*)pb;
        b[i].p.hi = *(const v8bf*)(pb + 16);
      }
      #pragma unroll
      for (int i = 0; i < 2; ++i)
        #pragma unroll
        for (int j = 0; j < 2; ++j)
          acc[i][j] = __builtin_amdgcn_wmma_f32_16x16x32_bf16(
              false, a[i].v, false, b[j].v, (short)0, acc[i][j], false, false);
    }

    wait_async0();        // our async loads into buf^1 are complete
    __syncthreads();      // everyone's are
    cur ^= 1;
  }

  // epilogue: C 16x16 f32 layout: lane<16 -> M=v, lane>=16 -> M=v+8; N=lane&15
  #pragma unroll
  for (int i = 0; i < 2; ++i) {
    #pragma unroll
    for (int j = 0; j < 2; ++j) {
      int col = n0 + wn * 32 + j * 16 + r16;
      float bv = bias ? bias[col] : 0.f;
      #pragma unroll
      for (int v = 0; v < 8; ++v) {
        int row = m0 + wm * 32 + i * 16 + v + half * 8;
        float val = acc[i][j][v] + bv;
        if (STORE_BF16)
          ((bf16*)Cout)[(size_t)row * N + col] = (bf16)val;
        else
          ((float*)Cout)[(size_t)row * N + col] = val;
      }
    }
  }
}

// ---------------------------------------------------------------------------
// Pointwise / reduction kernels
// ---------------------------------------------------------------------------
__global__ void zero_losses_kernel(float* out) {
  out[(size_t)M_TOT * V_ + 0] = 0.f;
  out[(size_t)M_TOT * V_ + 1] = 0.f;
}

__global__ __launch_bounds__(256)
void pack_bf16_kernel(const float* __restrict__ src, bf16* __restrict__ dst, long n) {
  long i = (long)blockIdx.x * 256 + threadIdx.x;
  if (i < n) dst[i] = (bf16)src[i];
}

// Wcell(1024x1024) = [[Wr, -Wi], [Wi, Wr]]  (row n, col k; all D=512 blocks)
__global__ __launch_bounds__(256)
void pack_wcell_kernel(const float* __restrict__ Wr, const float* __restrict__ Wi,
                       bf16* __restrict__ out) {
  long i = (long)blockIdx.x * 256 + threadIdx.x;   // over 1024*1024
  int n = (int)(i >> 10), k = (int)(i & 1023);
  float v;
  if (n < D_) v = (k < D_) ? Wr[(size_t)n * D_ + k] : -Wi[(size_t)n * D_ + (k - D_)];
  else        v = (k < D_) ? Wi[(size_t)(n - D_) * D_ + k] : Wr[(size_t)(n - D_) * D_ + (k - D_)];
  out[i] = (bf16)v;
}

__global__ __launch_bounds__(256)
void embed_kernel(const int* __restrict__ x, const float* __restrict__ mag,
                  const float* __restrict__ phase,
                  float* __restrict__ zr, float* __restrict__ zi, bf16* __restrict__ zf) {
  long i = (long)blockIdx.x * 256 + threadIdx.x;   // over M*D
  int m = (int)(i >> 9), d = (int)(i & 511);
  int tok = x[m];
  float r = mag[(size_t)tok * D_ + d], t = phase[(size_t)tok * D_ + d];
  float a = r * cosf(t), b = r * sinf(t);
  zr[i] = a; zi[i] = b;
  zf[(size_t)m * TWO_D + d]      = (bf16)a;
  zf[(size_t)m * TWO_D + D_ + d] = (bf16)b;
}

// L = [lr | li] (Mx1024, f32) -> magnitude norm + tanh -> zr, zi, zf(bf16)
__global__ __launch_bounds__(256)
void cell_post_kernel(const float* __restrict__ L,
                      float* __restrict__ zr, float* __restrict__ zi,
                      bf16* __restrict__ zf) {
  long i = (long)blockIdx.x * 256 + threadIdx.x;   // over M*D
  int m = (int)(i >> 9), d = (int)(i & 511);
  float lr = L[(size_t)m * TWO_D + d];
  float li = L[(size_t)m * TWO_D + D_ + d];
  float mm = sqrtf(lr * lr + li * li + EPS_);
  float inv = 1.f / (1.f + mm);
  float a = tanhf(lr * inv), b = tanhf(li * inv);
  zr[i] = a; zi[i] = b;
  zf[(size_t)m * TWO_D + d]      = (bf16)a;
  zf[(size_t)m * TWO_D + D_ + d] = (bf16)b;
}

// one wave per row: z2[m] = ||zr||^2 + ||zi||^2
__global__ __launch_bounds__(256)
void rownorm2_kernel(const float* __restrict__ zr, const float* __restrict__ zi,
                     float* __restrict__ z2) {
  int lane = threadIdx.x & 31, wid = threadIdx.x >> 5;
  int row = blockIdx.x * 8 + wid;
  float a = 0.f;
  for (int d = lane; d < D_; d += 32) {
    float x = zr[(size_t)row * D_ + d], y = zi[(size_t)row * D_ + d];
    a += x * x + y * y;
  }
  a = waveSum(a);
  if (lane == 0) z2[row] = a;
}

__global__ __launch_bounds__(256)
void booknorm_kernel(const float* __restrict__ book, float* __restrict__ b2, int rows) {
  int lane = threadIdx.x & 31, wid = threadIdx.x >> 5;
  int row = blockIdx.x * 8 + wid;
  if (row >= rows) return;
  float a = 0.f;
  for (int k = lane; k < TWO_D; k += 32) {
    float v = book[(size_t)row * TWO_D + k];
    a += v * v;
  }
  a = waveSum(a);
  if (lane == 0) b2[row] = a;
}

// attention: one wave per row; K_j/V_j precomputed in bf16 at append time
__global__ __launch_bounds__(256)
void attend_kernel(const float* __restrict__ Q, const bf16* __restrict__ Kbase,
                   const bf16* __restrict__ Vbase, int nm,
                   const float* __restrict__ conf,
                   float* __restrict__ zr, float* __restrict__ zi,
                   bf16* __restrict__ zf) {
  int lane = threadIdx.x & 31, wid = threadIdx.x >> 5;
  int row = blockIdx.x * 8 + wid;
  const float scale = 0.04419417382415922f;   // D^-0.5
  float cf = conf[row];
  float s[MEMSPAN_];
  float mx = -3.0e38f;
  for (int j = 0; j < nm; ++j) {
    const bf16* kp = Kbase + (size_t)j * M_TOT * D_ + (size_t)row * D_;
    const float* qp = Q + (size_t)row * D_;
    float part = 0.f;
    for (int k = lane; k < D_; k += 32) part += qp[k] * (float)kp[k];
    part = waveSum(part);
    float sc = part * scale * cf;
    s[j] = sc;
    mx = fmaxf(mx, sc);
  }
  float denom = 0.f;
  for (int j = 0; j < nm; ++j) { s[j] = __expf(s[j] - mx); denom += s[j]; }
  float inv = 1.f / denom;
  for (int e = lane; e < TWO_D; e += 32) {
    float ctx = 0.f;
    for (int j = 0; j < nm; ++j)
      ctx += s[j] * (float)Vbase[(size_t)j * M_TOT * TWO_D + (size_t)row * TWO_D + e];
    ctx *= 0.1f * inv;
    float nv;
    if (e < D_) { nv = zr[(size_t)row * D_ + e] + ctx;        zr[(size_t)row * D_ + e] = nv; }
    else        { nv = zi[(size_t)row * D_ + e - D_] + ctx;   zi[(size_t)row * D_ + e - D_] = nv; }
    zf[(size_t)row * TWO_D + e] = (bf16)nv;
  }
}

// per-row argmin over sym distances; also writes conf = 1/(1+dist)
__global__ __launch_bounds__(256)
void select_sym_kernel(const float* __restrict__ G, const float* __restrict__ b2,
                       const float* __restrict__ z2,
                       int* __restrict__ si, float* __restrict__ conf) {
  int lane = threadIdx.x & 31, wid = threadIdx.x >> 5;
  int row = blockIdx.x * 8 + wid;
  float bd = 3.4e38f; int bi = 0;
  for (int n = lane; n < NSYM_; n += 32) {
    float d = b2[n] - 2.f * G[(size_t)row * NSYM_ + n];
    if (d < bd) { bd = d; bi = n; }
  }
  waveArgMin(bd, bi);
  if (lane == 0) {
    si[row] = bi;
    float dist = z2[row] + bd;
    conf[row] = 1.f / (1.f + dist);
  }
}

// straight-through: state <- sym[si]; accumulate (1+CC)*mean((zq-z)^2)
__global__ __launch_bounds__(256)
void apply_sym_kernel(const int* __restrict__ si, const float* __restrict__ sym,
                      float* __restrict__ zr, float* __restrict__ zi,
                      bf16* __restrict__ zf, float* __restrict__ lossOut, float coef) {
  __shared__ float red[256];
  size_t idx = (size_t)blockIdx.x * 256 + threadIdx.x;  // over M*1024
  int m = (int)(idx >> 10), e = (int)(idx & 1023);
  float q = sym[(size_t)si[m] * TWO_D + e];
  float old;
  if (e < D_) { old = zr[(size_t)m * D_ + e];       zr[(size_t)m * D_ + e] = q; }
  else        { old = zi[(size_t)m * D_ + e - D_];  zi[(size_t)m * D_ + e - D_] = q; }
  zf[idx] = (bf16)q;
  float d = q - old;
  red[threadIdx.x] = d * d;
  __syncthreads();
  #pragma unroll
  for (int s = 128; s > 0; s >>= 1) {
    if (threadIdx.x < (unsigned)s) red[threadIdx.x] += red[threadIdx.x + s];
    __syncthreads();
  }
  if (threadIdx.x == 0) atomicAdd(lossOut, red[0] * coef);
}

// con quantize: argmin over 64 + loss vs current (already-sym) state
__global__ __launch_bounds__(256)
void quant_con_kernel(const float* __restrict__ G2, const float* __restrict__ b2c,
                      const float* __restrict__ con,
                      const float* __restrict__ zr, const float* __restrict__ zi,
                      float* __restrict__ lossOut, float coef) {
  int lane = threadIdx.x & 31, wid = threadIdx.x >> 5;
  int row = blockIdx.x * 8 + wid;
  float bd = 3.4e38f; int bi = 0;
  for (int n = lane; n < NCON_; n += 32) {
    float d = b2c[n] - 2.f * G2[(size_t)row * NCON_ + n];
    if (d < bd) { bd = d; bi = n; }
  }
  waveArgMin(bd, bi);       // all lanes now agree on bi
  float acc = 0.f;
  for (int e = lane; e < TWO_D; e += 32) {
    float zq = con[(size_t)bi * TWO_D + e];
    float zs = (e < D_) ? zr[(size_t)row * D_ + e] : zi[(size_t)row * D_ + e - D_];
    float df = zq - zs;
    acc += df * df;
  }
  acc = waveSum(acc);
  if (lane == 0) atomicAdd(lossOut, acc * coef);
}

// ---------------------------------------------------------------------------
// Host orchestration
// ---------------------------------------------------------------------------
extern "C" void kernel_launch(void* const* d_in, const int* in_sizes, int n_in,
                              void* d_out, int out_size, void* d_ws, size_t ws_size,
                              hipStream_t stream) {
  const int*   x     = (const int*)  d_in[0];
  const float* mag   = (const float*)d_in[1];
  const float* phase = (const float*)d_in[2];
  const float* Wr    = (const float*)d_in[3];
  const float* Wi    = (const float*)d_in[4];
  const float* qw    = (const float*)d_in[5];
  const float* qb    = (const float*)d_in[6];
  const float* kw    = (const float*)d_in[7];
  const float* kb    = (const float*)d_in[8];
  const float* vw    = (const float*)d_in[9];
  const float* vb    = (const float*)d_in[10];
  const float* dec_w = (const float*)d_in[11];
  const float* dec_b = (const float*)d_in[12];
  const float* sym   = (const float*)d_in[13];
  const float* con   = (const float*)d_in[14];
  float* out = (float*)d_out;
  float* symL = out + (size_t)M_TOT * V_;      // out, then symL, conL flat
  float* conL = symL + 1;

  // workspace carve-out (~470 MB)
  char* p = (char*)d_ws;
  auto alloc = [&](size_t bytes) -> char* {
    char* r = p; p += (bytes + 255) & ~(size_t)255; return r;
  };
  float* zr   = (float*)alloc((size_t)M_TOT * D_ * 4);
  float* zi   = (float*)alloc((size_t)M_TOT * D_ * 4);
  bf16*  zf   = (bf16*) alloc((size_t)M_TOT * TWO_D * 2);
  float* gbuf = (float*)alloc((size_t)M_TOT * TWO_D * 4);  // reused GEMM output
  float* z2   = (float*)alloc((size_t)M_TOT * 4);
  float* conf = (float*)alloc((size_t)M_TOT * 4);
  int*   si   = (int*)  alloc((size_t)M_TOT * 4);
  float* b2s  = (float*)alloc((size_t)NSYM_ * 4);
  float* b2c  = (float*)alloc((size_t)NCON_ * 4);
  bf16* wcell = (bf16*)alloc((size_t)TWO_D * TWO_D * 2);
  bf16* qwb   = (bf16*)alloc((size_t)D_ * TWO_D * 2);
  bf16* kwb   = (bf16*)alloc((size_t)D_ * TWO_D * 2);
  bf16* vwb   = (bf16*)alloc((size_t)TWO_D * TWO_D * 2);
  bf16* decwb = (bf16*)alloc((size_t)V_ * TWO_D * 2);
  bf16* symb  = (bf16*)alloc((size_t)NSYM_ * TWO_D * 2);
  bf16* conb  = (bf16*)alloc((size_t)NCON_ * TWO_D * 2);
  bf16* Kmem  = (bf16*)alloc((size_t)DEPTH_ * M_TOT * D_ * 2);
  bf16* Vmem  = (bf16*)alloc((size_t)DEPTH_ * M_TOT * TWO_D * 2);

  const float coef = (1.f + CC_) / (float)((size_t)M_TOT * TWO_D);
  const dim3 blk(256);
  const int gMD  = (M_TOT * D_) / 256;       // elementwise over M*D
  const int gM2D = (M_TOT * TWO_D) / 256;    // elementwise over M*2D
  const int gRow = M_TOT / 8;                // one wave per row

  auto gemmF = [&](const bf16* A, const bf16* W, const float* bias, float* C, int N) {
    gemm_bf16_kernel<false><<<dim3(N / TILE_N, M_TOT / TILE_M), blk, 0, stream>>>(
        A, W, bias, (void*)C, M_TOT, N, TWO_D);
  };
  auto gemmB = [&](const bf16* A, const bf16* W, const float* bias, bf16* C, int N) {
    gemm_bf16_kernel<true><<<dim3(N / TILE_N, M_TOT / TILE_M), blk, 0, stream>>>(
        A, W, bias, (void*)C, M_TOT, N, TWO_D);
  };

  // ---- prologue: zero losses, pack weights to bf16, codebook norms ----
  zero_losses_kernel<<<1, 1, 0, stream>>>(out);
  pack_wcell_kernel<<<(TWO_D * TWO_D) / 256, blk, 0, stream>>>(Wr, Wi, wcell);
  pack_bf16_kernel<<<(D_ * TWO_D) / 256, blk, 0, stream>>>(qw, qwb, (long)D_ * TWO_D);
  pack_bf16_kernel<<<(D_ * TWO_D) / 256, blk, 0, stream>>>(kw, kwb, (long)D_ * TWO_D);
  pack_bf16_kernel<<<(TWO_D * TWO_D) / 256, blk, 0, stream>>>(vw, vwb, (long)TWO_D * TWO_D);
  pack_bf16_kernel<<<(V_ * TWO_D) / 256, blk, 0, stream>>>(dec_w, decwb, (long)V_ * TWO_D);
  pack_bf16_kernel<<<(NSYM_ * TWO_D) / 256, blk, 0, stream>>>(sym, symb, (long)NSYM_ * TWO_D);
  pack_bf16_kernel<<<(NCON_ * TWO_D) / 256, blk, 0, stream>>>(con, conb, (long)NCON_ * TWO_D);
  booknorm_kernel<<<NSYM_ / 8, blk, 0, stream>>>(sym, b2s, NSYM_);
  booknorm_kernel<<<NCON_ / 8, blk, 0, stream>>>(con, b2c, NCON_);

  // ---- embedding ----
  embed_kernel<<<gMD, blk, 0, stream>>>(x, mag, phase, zr, zi, zf);

  // ---- depth loop ----
  for (int it = 0; it < DEPTH_; ++it) {
    // cell: [lr|li] = [zr|zi] @ Wcell^T, then norm+tanh
    gemmF(zf, wcell, nullptr, gbuf, TWO_D);
    cell_post_kernel<<<gMD, blk, 0, stream>>>(gbuf, zr, zi, zf);

    if (it > 0) {  // attend over cached K/V (nm = it slots), conf from prev iter
      gemmF(zf, qwb, qb, gbuf, D_);                 // Q (Mx512) into gbuf
      attend_kernel<<<gRow, blk, 0, stream>>>(gbuf, Kmem, Vmem, it, conf, zr, zi, zf);
    }

    // sym quantize: dist = z2 + b2 - 2*G, straight-through + loss + conf
    rownorm2_kernel<<<gRow, blk, 0, stream>>>(zr, zi, z2);
    gemmF(zf, symb, nullptr, gbuf, NSYM_);          // G (Mx512)
    select_sym_kernel<<<gRow, blk, 0, stream>>>(gbuf, b2s, z2, si, conf);
    apply_sym_kernel<<<gM2D, blk, 0, stream>>>(si, sym, zr, zi, zf, symL, coef);

    // con quantize: loss only (state stays at sym rows)
    gemmF(zf, conb, nullptr, gbuf, NCON_);          // G2 (Mx64)
    quant_con_kernel<<<gRow, blk, 0, stream>>>(gbuf, b2c, con, zr, zi, conL, coef);

    // append to memory: precompute K_m, V_m once (kw/vw invariant)
    gemmB(zf, kwb, kb, Kmem + (size_t)it * M_TOT * D_,    D_);
    gemmB(zf, vwb, vb, Vmem + (size_t)it * M_TOT * TWO_D, TWO_D);
  }

  // ---- lookahead cells ----
  for (int l = 0; l < LOOK_; ++l) {
    gemmF(zf, wcell, nullptr, gbuf, TWO_D);
    cell_post_kernel<<<gMD, blk, 0, stream>>>(gbuf, zr, zi, zf);
  }

  // ---- decoder ----
  gemmF(zf, decwb, dec_b, out, V_);
}
